// GLABlock_76261439308440
// MI455X (gfx1250) — matrix operand order repaired and checked by reading, not compiled, not run on hardware
//
#include <hip/hip_runtime.h>
#include <hip/hip_bf16.h>

// ---------------------------------------------------------------------------
// Types for CDNA5 WMMA (wave32, 16x16x32 bf16 -> f32)
// ---------------------------------------------------------------------------
typedef __attribute__((ext_vector_type(16))) __bf16       v16bf;
typedef __attribute__((ext_vector_type(8)))  float        v8f;
typedef __attribute__((ext_vector_type(4)))  unsigned int v4u;

#define BT   4096      // B*T
#define DMODEL 2048
#define KDIM 1024
#define VDIM 2048
#define HIDDEN 5504
#define NH   8
#define HK   128
#define HV   256
#define CHUNK 64
#define NCHUNK 32

// ---------------------------------------------------------------------------
// gfx1250 async global->LDS copy (ASYNCcnt-tracked, bypasses VGPRs).
// LDS address taken from the vdst VGPR; 16 bytes per lane per instruction.
// ---------------------------------------------------------------------------
__device__ __forceinline__ void async_copy16(unsigned lds_off, const void* gptr) {
  asm volatile("global_load_async_to_lds_b128 %0, %1, off"
               :: "v"(lds_off), "v"(gptr) : "memory");
}
__device__ __forceinline__ void wait_async0() {
  asm volatile("s_wait_asynccnt 0" ::: "memory");
}

// ---------------------------------------------------------------------------
// Transposing f32 -> bf16 weight convert: out[n*K + k] = (bf16) in[k*N + n].
// Done once per launch so the GEMM hot loop never transposes.
// 64x64 tile per block via LDS.
// ---------------------------------------------------------------------------
__global__ __launch_bounds__(256) void k_conv_transpose_bf16(
    const float* __restrict__ in, __bf16* __restrict__ out, int K, int N) {
  __shared__ float t[64][65];
  int tk0 = blockIdx.y * 64, tn0 = blockIdx.x * 64;
  int tr  = threadIdx.x >> 2;         // 0..63
  int tc4 = (threadIdx.x & 3) * 16;   // 16-wide strip
  const float* src = in + (size_t)(tk0 + tr) * N + tn0 + tc4;
#pragma unroll
  for (int e = 0; e < 16; e++) t[tr][tc4 + e] = src[e];
  __syncthreads();
  __bf16* dst = out + (size_t)(tn0 + tr) * K + tk0 + tc4;
#pragma unroll
  for (int e = 0; e < 16; e++) dst[e] = (__bf16)t[tc4 + e][tr];
}

// ---------------------------------------------------------------------------
// RMSNorm over last dim, output bf16 (feeds WMMA GEMMs)
// ---------------------------------------------------------------------------
__global__ __launch_bounds__(256) void k_rmsnorm_bf16(
    const float* __restrict__ x, const float* __restrict__ w,
    __bf16* __restrict__ out, int cols) {
  int row = blockIdx.x;
  const float* xr = x + (size_t)row * cols;
  __shared__ float red[256];
  float s = 0.f;
  for (int c = threadIdx.x; c < cols; c += 256) { float v = xr[c]; s += v * v; }
  red[threadIdx.x] = s;
  __syncthreads();
  for (int off = 128; off > 0; off >>= 1) {
    if ((int)threadIdx.x < off) red[threadIdx.x] += red[threadIdx.x + off];
    __syncthreads();
  }
  float inv = rsqrtf(red[0] / (float)cols + 1e-6f);
  __bf16* orow = out + (size_t)row * cols;
  for (int c = threadIdx.x; c < cols; c += 256)
    orow[c] = (__bf16)(xr[c] * inv * w[c]);
}

// ---------------------------------------------------------------------------
// WMMA GEMM: C[M,N](f32) = A[M,K](bf16 row-major) @ Bt[N,K](bf16, B transposed)
// Block = 256 threads = 8 waves (wave32). Tile BM=BN=128, BK=32.
// Double-buffered LDS tiles filled by async global->LDS copies; WMMAs on
// tile k overlap the async fill of tile k+1.
// ---------------------------------------------------------------------------
#define LDT_S 40   // 32 + pad (bf16 elems); 80B rows keep 16B alignment

__global__ __launch_bounds__(256) void k_gemm_bf16(
    const __bf16* __restrict__ A, const __bf16* __restrict__ Bt,
    float* __restrict__ C, int M, int N, int K) {
  __shared__ __bf16 As[2][128 * LDT_S];   // [row][k]
  __shared__ __bf16 Bs[2][128 * LDT_S];   // [col][k]

  int tid  = threadIdx.x;
  int wave = tid >> 5;
  int lane = tid & 31;
  int wm   = wave & 1;        // row group (64 rows)
  int wn   = wave >> 1;       // col group (32 cols)
  int row0 = blockIdx.y * 128;
  int col0 = blockIdx.x * 128;

  // per-thread staging slot: 16+16 bf16 of one tile row
  int r   = tid >> 1;
  int seg = (tid & 1) * 16;
  const __bf16* arow = A  + (size_t)(row0 + r) * K + seg;
  const __bf16* brow = Bt + (size_t)(col0 + r) * K + seg;
  unsigned ldsA[2], ldsB[2];
#pragma unroll
  for (int buf = 0; buf < 2; buf++) {
    ldsA[buf] = (unsigned)(size_t)&As[buf][r * LDT_S + seg];
    ldsB[buf] = (unsigned)(size_t)&Bs[buf][r * LDT_S + seg];
  }

  // preload tile 0 into buffer 0
  async_copy16(ldsA[0],      arow);
  async_copy16(ldsA[0] + 16, arow + 8);
  async_copy16(ldsB[0],      brow);
  async_copy16(ldsB[0] + 16, brow + 8);
  wait_async0();
  __syncthreads();

  v8f acc[4][2];
#pragma unroll
  for (int i = 0; i < 4; i++)
#pragma unroll
    for (int j = 0; j < 2; j++)
#pragma unroll
      for (int v = 0; v < 8; v++) acc[i][j][v] = 0.f;

  union Frag { v4u u[2]; v16bf v; };

  int m   = lane & 15;
  int hi  = lane >> 4;
  int kbA = hi * 16;   // A frag: lanes 0-15 -> K 0..15, lanes 16-31 -> K 16..31
  int kbB = hi * 16;   // B frag: lanes 0-15 -> K 0..15, lanes 16-31 -> K 16..31

  for (int k0 = 0; k0 < K; k0 += 32) {
    int cur = (k0 >> 5) & 1, nxt = cur ^ 1;

    // async-fill next tile while computing on current
    if (k0 + 32 < K) {
      const __bf16* an = arow + k0 + 32;
      const __bf16* bn = brow + k0 + 32;
      async_copy16(ldsA[nxt],      an);
      async_copy16(ldsA[nxt] + 16, an + 8);
      async_copy16(ldsB[nxt],      bn);
      async_copy16(ldsB[nxt] + 16, bn + 8);
    }

    v16bf afr[4];
#pragma unroll
    for (int i = 0; i < 4; i++) {
      Frag f;
      const __bf16* p = &As[cur][(wm * 64 + i * 16 + m) * LDT_S + kbA];
      f.u[0] = *(const v4u*)p;          // elems 0..7  : K kbA..kbA+7
      f.u[1] = *(const v4u*)(p + 8);    // elems 8..15 : K kbA+8..kbA+15
      afr[i] = f.v;
    }
    v16bf bfr[2];
#pragma unroll
    for (int j = 0; j < 2; j++) {
      Frag f;
      const __bf16* p = &Bs[cur][(wn * 32 + j * 16 + m) * LDT_S + kbB];
      f.u[0] = *(const v4u*)p;          // K kbB..kbB+7
      f.u[1] = *(const v4u*)(p + 8);    // K kbB+8..kbB+15
      bfr[j] = f.v;
    }

#pragma unroll
    for (int i = 0; i < 4; i++)
#pragma unroll
      for (int j = 0; j < 2; j++)
        acc[i][j] = __builtin_amdgcn_wmma_f32_16x16x32_bf16(
            false, afr[i], false, bfr[j], (short)0, acc[i][j], false, false);

    wait_async0();     // own async fills of buf[nxt] landed
    __syncthreads();   // everyone done reading buf[cur]
  }

  // store: C/D layout -> row = v + 8*(lane>=16), col = lane&15
  int n    = lane & 15;
  int rofs = (lane >> 4) * 8;
#pragma unroll
  for (int i = 0; i < 4; i++)
#pragma unroll
    for (int j = 0; j < 2; j++) {
      float* cp = C + (size_t)(row0 + wm * 64 + i * 16 + rofs) * N +
                  col0 + wn * 32 + j * 16 + n;
#pragma unroll
      for (int v = 0; v < 8; v++) cp[(size_t)v * N] = acc[i][j][v];
    }
}

// ---------------------------------------------------------------------------
// Low-rank gate path stage 1: tmp[BT,16] = h_bf16[BT,D] @ gk_w1[D,16]
// ---------------------------------------------------------------------------
__global__ __launch_bounds__(256) void k_lowrank1(
    const __bf16* __restrict__ h, const float* __restrict__ w1,
    float* __restrict__ tmp, int K) {
  int row = blockIdx.x;
  int col = threadIdx.x & 15;
  int sl  = threadIdx.x >> 4;   // 16 K-slices
  const __bf16* hr = h + (size_t)row * K;
  int kps = K / 16;
  float acc = 0.f;
  for (int k = sl * kps; k < (sl + 1) * kps; k++)
    acc += (float)hr[k] * w1[(size_t)k * 16 + col];
  __shared__ float red[256];
  red[threadIdx.x] = acc;
  __syncthreads();
  if (sl == 0) {
    float t = 0.f;
    for (int s2 = 0; s2 < 16; s2++) t += red[s2 * 16 + col];
    tmp[(size_t)row * 16 + col] = t;
  }
}

// ---------------------------------------------------------------------------
// Stage 2: g = log_sigmoid(tmp @ gk_w2 + gk_b) / GATE_NORM   (K=16)
// ---------------------------------------------------------------------------
__global__ __launch_bounds__(256) void k_lowrank2_gate(
    const float* __restrict__ tmp, const float* __restrict__ w2,
    const float* __restrict__ bias, float* __restrict__ g, int N) {
  size_t idx = (size_t)blockIdx.x * 256 + threadIdx.x;
  int row = (int)(idx / N), col = (int)(idx % N);
  const float* tr = tmp + (size_t)row * 16;
  float y = bias[col];
#pragma unroll
  for (int r = 0; r < 16; r++) y += tr[r] * w2[(size_t)r * N + col];
  float ls = fminf(y, 0.f) - log1pf(__expf(-fabsf(y)));   // stable log_sigmoid
  g[idx] = ls * (1.0f / 16.0f);                            // / GATE_NORM
}

// ---------------------------------------------------------------------------
// GLA chunked scan. One workgroup per (b,h); state S (128x256 f32 = 128KB)
// resident in LDS across all 32 chunks. Dynamic LDS layout (floats):
//   S[128*256] | qe[64*128] | kd[64*128] | bs[64*128] | kbl[64*128]
//   | A[64*64] | bl[128]        (v chunk overlays kd+bs once both are dead)
// v chunk is staged with async global->LDS copies.
// ---------------------------------------------------------------------------
__global__ __launch_bounds__(256) void k_gla(
    const float* __restrict__ q, const float* __restrict__ k,
    const float* __restrict__ v, const float* __restrict__ g,
    float* __restrict__ o) {
  extern __shared__ float sm[];
  float* S    = sm;                    // 32768
  float* qe   = S   + 128 * 256;       // 8192
  float* kd   = qe  + 64 * 128;        // 8192
  float* bs   = kd  + 64 * 128;        // 8192
  float* kbl  = bs  + 64 * 128;        // 8192
  float* Am   = kbl + 64 * 128;        // 4096
  float* bl   = Am  + 64 * 64;         // 128
  float* vbuf = kd;                    // overlays kd+bs : 64*256 floats

  int bh  = blockIdx.x;                // 0..15
  int b   = bh >> 3, h = bh & 7;
  int tid = threadIdx.x;
  const size_t qs = KDIM, vs = VDIM;
  const float scale = 0.08838834764831845f;   // HK^-0.5

  for (int i = tid; i < 128 * 256; i += 256) S[i] = 0.f;
  __syncthreads();

  for (int c = 0; c < NCHUNK; c++) {
    size_t rowbase = (size_t)b * 2048 + (size_t)c * CHUNK;

    // 1: per-k-column cumulative decay b = cumsum(g)
    if (tid < HK) {
      float run = 0.f;
      for (int i = 0; i < CHUNK; i++) {
        run += g[(rowbase + i) * qs + (size_t)h * HK + tid];
        bs[i * HK + tid] = run;
      }
      bl[tid] = run;
    }
    __syncthreads();

    // 2: qe = q*exp(b)*scale, kd = k*exp(-b), kbl = k*exp(bl-b)
    for (int e = tid; e < CHUNK * HK; e += 256) {
      int i = e >> 7, kk = e & 127;
      float bv  = bs[e];
      float blv = bl[kk];
      size_t gi = (rowbase + i) * qs + (size_t)h * HK + kk;
      float qv = q[gi], kv = k[gi];
      qe[e]  = qv * __expf(bv) * scale;
      kd[e]  = kv * __expf(-bv);
      kbl[e] = kv * __expf(blv - bv);
    }
    __syncthreads();

    // 3: A = tril(qe @ kd^T)   64x64
    for (int e = tid; e < CHUNK * CHUNK; e += 256) {
      int i = e >> 6, j = e & 63;
      float s = 0.f;
      if (j <= i) {
#pragma unroll 8
        for (int kk = 0; kk < HK; kk++) s += qe[i * HK + kk] * kd[j * HK + kk];
      }
      Am[e] = s;
    }
    __syncthreads();

    // stage v chunk into LDS via async copies (kd & bs are dead now)
    {
      const float* vsrc = v + rowbase * vs + (size_t)h * HV;
      for (int e = tid * 4; e < CHUNK * HV; e += 256 * 4) {
        int i = e >> 8, cc = e & 255;
        async_copy16((unsigned)(size_t)&vbuf[e], vsrc + (size_t)i * vs + cc);
      }
      wait_async0();
    }
    __syncthreads();

    // 4: o = qe @ S + A @ v   (thread owns row i, 64-wide col strip)
    {
      int i  = tid >> 2;
      int cs = (tid & 3) * 64;
      float accv[64];
#pragma unroll
      for (int cc = 0; cc < 64; cc++) accv[cc] = 0.f;
      for (int kk = 0; kk < HK; kk++) {
        float qv = qe[i * HK + kk];
        const float* Sr = &S[kk * HV + cs];
#pragma unroll
        for (int cc = 0; cc < 64; cc++) accv[cc] += qv * Sr[cc];
      }
      for (int j = 0; j <= i; j++) {
        float av = Am[i * 64 + j];
        const float* vr = &vbuf[j * HV + cs];
#pragma unroll
        for (int cc = 0; cc < 64; cc++) accv[cc] += av * vr[cc];
      }
      float* orow = o + (rowbase + i) * vs + (size_t)h * HV + cs;
#pragma unroll
      for (int cc = 0; cc < 64; cc++) orow[cc] = accv[cc];
    }
    __syncthreads();

    // 5: S = exp(bl) (.) S + kbl^T @ v
    for (int half = 0; half < 2; half++) {
      int kk = (tid >> 2) + half * 64;
      int cs = (tid & 3) * 64;
      float dec = __expf(bl[kk]);
      float accs[64];
      const float* Sr0 = &S[kk * HV + cs];
#pragma unroll
      for (int cc = 0; cc < 64; cc++) accs[cc] = dec * Sr0[cc];
      for (int j = 0; j < CHUNK; j++) {
        float kv = kbl[j * HK + kk];
        const float* vr = &vbuf[j * HV + cs];
#pragma unroll
        for (int cc = 0; cc < 64; cc++) accs[cc] += kv * vr[cc];
      }
      float* Sw = &S[kk * HV + cs];
#pragma unroll
      for (int cc = 0; cc < 64; cc++) Sw[cc] = accs[cc];
    }
    __syncthreads();
  }
}

// ---------------------------------------------------------------------------
// Per-(token,head) RMSNorm + SiLU-gate; output bf16 for the WO GEMM.
// ---------------------------------------------------------------------------
__global__ __launch_bounds__(256) void k_gate_rms(
    const float* __restrict__ o, const float* __restrict__ gate,
    const float* __restrict__ gw, __bf16* __restrict__ outb) {
  int row = blockIdx.x;
  int c   = threadIdx.x;
  size_t base = (size_t)row * HV;
  float ov = o[base + c];
  __shared__ float red[256];
  red[c] = ov * ov;
  __syncthreads();
  for (int off = 128; off > 0; off >>= 1) {
    if (c < off) red[c] += red[c + off];
    __syncthreads();
  }
  float inv = rsqrtf(red[0] * (1.0f / HV) + 1e-6f);
  float gv  = gate[base + c];
  float sig = 1.f / (1.f + __expf(-gv));
  outb[base + c] = (__bf16)(ov * inv * gw[c] * gv * sig);
}

// ---------------------------------------------------------------------------
// Elementwise helpers
// ---------------------------------------------------------------------------
__global__ void k_add(const float* __restrict__ a, const float* __restrict__ b,
                      float* __restrict__ out, size_t n) {
  size_t i = (size_t)blockIdx.x * blockDim.x + threadIdx.x;
  size_t stride = (size_t)gridDim.x * blockDim.x;
  for (; i < n; i += stride) out[i] = a[i] + b[i];
}

__global__ void k_silu_mul_bf16(const float* __restrict__ g1,
                                const float* __restrict__ u,
                                __bf16* __restrict__ f, size_t n) {
  size_t i = (size_t)blockIdx.x * blockDim.x + threadIdx.x;
  size_t stride = (size_t)gridDim.x * blockDim.x;
  for (; i < n; i += stride) {
    float x = g1[i];
    float s = x / (1.f + __expf(-x));
    f[i] = (__bf16)(s * u[i]);
  }
}

// ---------------------------------------------------------------------------
// Host orchestration
// ---------------------------------------------------------------------------
extern "C" void kernel_launch(void* const* d_in, const int* in_sizes, int n_in,
                              void* d_out, int out_size, void* d_ws, size_t ws_size,
                              hipStream_t stream) {
  const float* x      = (const float*)d_in[0];
  const float* norm1w = (const float*)d_in[1];
  const float* wq     = (const float*)d_in[2];
  const float* wk     = (const float*)d_in[3];
  const float* wv     = (const float*)d_in[4];
  const float* gkw1   = (const float*)d_in[5];
  const float* gkw2   = (const float*)d_in[6];
  const float* gkb    = (const float*)d_in[7];
  const float* wg     = (const float*)d_in[8];
  const float* gnormw = (const float*)d_in[9];
  const float* wo     = (const float*)d_in[10];
  const float* ffnw   = (const float*)d_in[11];
  const float* wgate  = (const float*)d_in[12];
  const float* wup    = (const float*)d_in[13];
  const float* wdown  = (const float*)d_in[14];
  float* out = (float*)d_out;

  char* ws = (char*)d_ws;
  size_t cur = 0;
  auto alloc = [&](size_t bytes) {
    size_t off = cur;
    cur = (cur + bytes + 255) & ~(size_t)255;
    return off;
  };

  // bf16 transposed weight copies (Bt layout [N][K])
  __bf16* wq_t  = (__bf16*)(ws + alloc((size_t)DMODEL * KDIM * 2));
  __bf16* wk_t  = (__bf16*)(ws + alloc((size_t)DMODEL * KDIM * 2));
  __bf16* wv_t  = (__bf16*)(ws + alloc((size_t)DMODEL * VDIM * 2));
  __bf16* wg_t  = (__bf16*)(ws + alloc((size_t)DMODEL * VDIM * 2));
  __bf16* wo_t  = (__bf16*)(ws + alloc((size_t)VDIM * DMODEL * 2));
  __bf16* wga_t = (__bf16*)(ws + alloc((size_t)DMODEL * HIDDEN * 2));
  __bf16* wu_t  = (__bf16*)(ws + alloc((size_t)DMODEL * HIDDEN * 2));
  __bf16* wd_t  = (__bf16*)(ws + alloc((size_t)HIDDEN * DMODEL * 2));
  // activations
  __bf16* h_b   = (__bf16*)(ws + alloc((size_t)BT * DMODEL * 2));
  float*  q_f   = (float*) (ws + alloc((size_t)BT * KDIM * 4));
  float*  k_f   = (float*) (ws + alloc((size_t)BT * KDIM * 4));
  float*  v_f   = (float*) (ws + alloc((size_t)BT * VDIM * 4));
  float*  g_f   = (float*) (ws + alloc((size_t)BT * KDIM * 4));
  float*  tmp16 = (float*) (ws + alloc((size_t)BT * 16 * 4));
  float*  gate_f= (float*) (ws + alloc((size_t)BT * VDIM * 4));
  float*  o_f   = (float*) (ws + alloc((size_t)BT * VDIM * 4));
  __bf16* o_b   = (__bf16*)(ws + alloc((size_t)BT * VDIM * 2));
  float*  attn_f= (float*) (ws + alloc((size_t)BT * DMODEL * 4));
  float*  x2_f  = (float*) (ws + alloc((size_t)BT * DMODEL * 4));
  __bf16* h2_b  = (__bf16*)(ws + alloc((size_t)BT * DMODEL * 2));
  float*  g1_f  = (float*) (ws + alloc((size_t)BT * HIDDEN * 4));
  float*  u_f   = (float*) (ws + alloc((size_t)BT * HIDDEN * 4));
  __bf16* f_b   = (__bf16*)(ws + alloc((size_t)BT * HIDDEN * 2));
  float*  ffn_f = (float*) (ws + alloc((size_t)BT * DMODEL * 4));

  auto convT = [&](const float* src, __bf16* dst, int K, int N) {
    k_conv_transpose_bf16<<<dim3(N / 64, K / 64), 256, 0, stream>>>(src, dst, K, N);
  };
  auto gemm = [&](const __bf16* A, const __bf16* Bt, float* C, int M, int N, int K) {
    k_gemm_bf16<<<dim3(N / 128, M / 128), 256, 0, stream>>>(A, Bt, C, M, N, K);
  };

  // 0: transpose+convert weights once per launch
  convT(wq,    wq_t,  DMODEL, KDIM);
  convT(wk,    wk_t,  DMODEL, KDIM);
  convT(wv,    wv_t,  DMODEL, VDIM);
  convT(wg,    wg_t,  DMODEL, VDIM);
  convT(wo,    wo_t,  VDIM,   DMODEL);
  convT(wgate, wga_t, DMODEL, HIDDEN);
  convT(wup,   wu_t,  DMODEL, HIDDEN);
  convT(wdown, wd_t,  HIDDEN, DMODEL);

  // 1: h = rmsnorm(x) -> bf16
  k_rmsnorm_bf16<<<BT, 256, 0, stream>>>(x, norm1w, h_b, DMODEL);

  // 2: projections (WMMA)
  gemm(h_b, wq_t, q_f,    BT, KDIM, DMODEL);
  gemm(h_b, wk_t, k_f,    BT, KDIM, DMODEL);
  gemm(h_b, wv_t, v_f,    BT, VDIM, DMODEL);
  gemm(h_b, wg_t, gate_f, BT, VDIM, DMODEL);

  // 3: low-rank decay gates
  k_lowrank1<<<BT, 256, 0, stream>>>(h_b, gkw1, tmp16, DMODEL);
  k_lowrank2_gate<<<(BT * KDIM) / 256, 256, 0, stream>>>(tmp16, gkw2, gkb, g_f, KDIM);

  // 4: chunked GLA scan (state in LDS)
  size_t gla_lds = (size_t)(128 * 256 + 4 * 64 * 128 + 64 * 64 + 128) * sizeof(float);
  k_gla<<<2 * NH, 256, gla_lds, stream>>>(q_f, k_f, v_f, g_f, o_f);

  // 5: per-head rmsnorm * silu-gate -> bf16
  k_gate_rms<<<BT * NH, 256, 0, stream>>>(o_f, gate_f, gnormw, o_b);

  // 6: output projection + residual
  gemm(o_b, wo_t, attn_f, BT, DMODEL, VDIM);
  k_add<<<4096, 256, 0, stream>>>(x, attn_f, x2_f, (size_t)BT * DMODEL);

  // 7: FFN
  k_rmsnorm_bf16<<<BT, 256, 0, stream>>>(x2_f, ffnw, h2_b, DMODEL);
  gemm(h2_b, wga_t, g1_f, BT, HIDDEN, DMODEL);
  gemm(h2_b, wu_t,  u_f,  BT, HIDDEN, DMODEL);
  k_silu_mul_bf16<<<8192, 256, 0, stream>>>(g1_f, u_f, f_b, (size_t)BT * HIDDEN);
  gemm(f_b, wd_t, ffn_f, BT, DMODEL, HIDDEN);
  k_add<<<4096, 256, 0, stream>>>(x2_f, ffn_f, out, (size_t)BT * DMODEL);
}
